// RiskSensitiveGNN_34454227648744
// MI455X (gfx1250) — matrix operand order repaired and checked by reading, not compiled, not run on hardware
//
#include <hip/hip_runtime.h>

#define NN      100000
#define NE      1600000
#define HID     64
#define HEADS   4
#define EHID    32
#define NLAYERS 3
#define LN_EPS  1e-5f

typedef __attribute__((ext_vector_type(16))) _Float16 v16h;
typedef __attribute__((ext_vector_type(8)))  _Float16 v8h;
typedef __attribute__((ext_vector_type(4)))  _Float16 v4h;
typedef __attribute__((ext_vector_type(8)))  float    v8f;

// ---------------- monotone float <-> u32 map for atomic max on floats ----------
__device__ __forceinline__ unsigned fmap_u(float f) {
  int i = __float_as_int(f);
  return (i >= 0) ? ((unsigned)i | 0x80000000u) : ~(unsigned)i;
}
__device__ __forceinline__ float funmap_u(unsigned u) {
  int i = (u & 0x80000000u) ? (int)(u & 0x7fffffffu) : (int)~u;
  return __int_as_float(i);
}

// ---------------- fold attention vectors into the linear weights ---------------
// Asw[i][k][hd] = sum_c lin[i,k,hd*64+c] * att_src[i,hd,c]   (same for dst)
// Aew[i][k][hd] = sum_c lin_edge[i,k,hd*64+c] * att_edge[i,hd,c]
__global__ void precompute_att(const float* __restrict__ lin,
                               const float* __restrict__ lin_edge,
                               const float* __restrict__ att_src,
                               const float* __restrict__ att_dst,
                               const float* __restrict__ att_edge,
                               float* __restrict__ Asw, float* __restrict__ Adw,
                               float* __restrict__ Aew) {
  for (int t = threadIdx.x; t < NLAYERS * HID * HEADS; t += blockDim.x) {
    int i = t >> 8; int r = t & 255; int k = r >> 2; int hd = r & 3;
    const float* lp = lin + ((size_t)i * HID + k) * (HEADS * HID) + hd * HID;
    const float* as = att_src + (i * HEADS + hd) * HID;
    const float* ad = att_dst + (i * HEADS + hd) * HID;
    float s = 0.f, d = 0.f;
    for (int c = 0; c < HID; ++c) { float lv = lp[c]; s += lv * as[c]; d += lv * ad[c]; }
    Asw[t] = s; Adw[t] = d;
  }
  for (int t = threadIdx.x; t < NLAYERS * EHID * HEADS; t += blockDim.x) {
    int i = t >> 7; int r = t & 127; int k = r >> 2; int hd = r & 3;
    const float* lp = lin_edge + ((size_t)i * EHID + k) * (HEADS * HID) + hd * HID;
    const float* ae = att_edge + (i * HEADS + hd) * HID;
    float s = 0.f;
    for (int c = 0; c < HID; ++c) s += lp[c] * ae[c];
    Aew[t] = s;
  }
}

// ---------------- pack W[K,N] (f32 row-major) into WMMA B fragments (f16) ------
// flat = ((kt*Ntiles + nt)*32 + lane)*16 + i ;  K = kt*32 + (lane>=16)*16 + i,
// N = nt*16 + (lane&15)   (contiguous k order, K halves split by lane group; A is loaded the same way)
__global__ void pack_b_f16(const float* __restrict__ W, _Float16* __restrict__ Bp,
                           int K, int N) {
  int tid = blockIdx.x * blockDim.x + threadIdx.x;
  if (tid >= K * N) return;
  int i    = tid & 15;
  int lane = (tid >> 4) & 31;
  int ntk  = tid >> 9;
  int Ntiles = N >> 4;
  int nt = ntk % Ntiles;
  int kt = ntk / Ntiles;
  int Kidx = kt * 32 + (lane >> 4) * 16 + i;
  int Nidx = nt * 16 + (lane & 15);
  Bp[tid] = (_Float16)W[(size_t)Kidx * N + Nidx];
}

// ---------------- WMMA GEMM: C[M,N] = A[M,K](f16 row-major) x Bp(packed) -------
// One wave -> one 16-row M tile x four 16-col N tiles. 8 waves / block.
__global__ void gemm_wmma_f16(const _Float16* __restrict__ A,
                              const _Float16* __restrict__ Bp,
                              float* __restrict__ C,
                              int Mtiles, int Ktiles, int Ntiles) {
  const int lane = threadIdx.x & 31;
  const int wave = threadIdx.x >> 5;
  const int w    = blockIdx.x * 8 + wave;
  const int ntg  = Ntiles >> 2;
  if (w >= Mtiles * ntg) return;
  const int mt = w / ntg;
  const int n0 = (w % ntg) * 4;          // first n-tile of this wave
  const int K  = Ktiles * 32;
  const int N  = Ntiles * 16;
  const int m    = mt * 16 + (lane & 15);
  const int koff = (lane >> 4) * 16;     // A layout: upper lane half gets K+16

  v8f acc[4] = {};
  for (int kt = 0; kt < Ktiles; ++kt) {
    union { v16h v; v8h h[2]; } a;
    const _Float16* ar = A + (size_t)m * K + kt * 32 + koff;
    a.h[0] = *(const v8h*)(ar);          // K = kt*32 + koff + [0..7]
    a.h[1] = *(const v8h*)(ar + 8);      // K = kt*32 + koff + 8 + [0..7]
    const _Float16* bbase = Bp + (((size_t)kt * Ntiles + n0) * 32 + lane) * 16;
#pragma unroll
    for (int t = 0; t < 4; ++t) {
      union { v16h v; v8h h[2]; } b;
      const _Float16* br = bbase + (size_t)t * 32 * 16;
      b.h[0] = *(const v8h*)(br);
      b.h[1] = *(const v8h*)(br + 8);
      acc[t] = __builtin_amdgcn_wmma_f32_16x16x32_f16(
          false, a.v, false, b.v, (short)0, acc[t], false, false);
    }
  }
  // C/D layout: VGPR v holds row M = v (+8 for lanes 16-31), col = lane&15
  const int row0 = mt * 16 + ((lane >> 4) ? 8 : 0);
  const int col  = lane & 15;
#pragma unroll
  for (int t = 0; t < 4; ++t) {
    float* cp = C + (size_t)row0 * N + (size_t)(n0 + t) * 16 + col;
#pragma unroll
    for (int v = 0; v < 8; ++v) cp[(size_t)v * N] = acc[t][v];
  }
}

// ---------------- node encoder: 9 -> 32 relu -> 64 -> LN -----------------------
__global__ void node_encoder(const float* __restrict__ x,
                             const float* __restrict__ w1, const float* __restrict__ b1,
                             const float* __restrict__ w2, const float* __restrict__ b2,
                             const float* __restrict__ g,  const float* __restrict__ beta,
                             float* __restrict__ h) {
  __shared__ float sw1[9 * 32], sb1[32], sw2[32 * 64], sb2[64], sg[64], sbt[64];
  for (int t = threadIdx.x; t < 288;  t += 256) sw1[t] = w1[t];
  for (int t = threadIdx.x; t < 2048; t += 256) sw2[t] = w2[t];
  if (threadIdx.x < 32) sb1[threadIdx.x] = b1[threadIdx.x];
  if (threadIdx.x < 64) { sb2[threadIdx.x] = b2[threadIdx.x];
                          sg[threadIdx.x]  = g[threadIdx.x];
                          sbt[threadIdx.x] = beta[threadIdx.x]; }
  __syncthreads();
  int n = blockIdx.x * 256 + threadIdx.x;
  if (n >= NN) return;
  float xi[9];
#pragma unroll
  for (int j = 0; j < 9; ++j) xi[j] = x[(size_t)n * 9 + j];
  float t1[32];
#pragma unroll
  for (int k = 0; k < 32; ++k) {
    float a = sb1[k];
#pragma unroll
    for (int j = 0; j < 9; ++j) a += xi[j] * sw1[j * 32 + k];
    t1[k] = a > 0.f ? a : 0.f;
  }
  float t2[64];
#pragma unroll
  for (int c = 0; c < 64; ++c) {
    float a = sb2[c];
#pragma unroll
    for (int k = 0; k < 32; ++k) a += t1[k] * sw2[k * 64 + c];
    t2[c] = a;
  }
  float mu = 0.f;
#pragma unroll
  for (int c = 0; c < 64; ++c) mu += t2[c];
  mu *= (1.f / 64.f);
  float var = 0.f;
#pragma unroll
  for (int c = 0; c < 64; ++c) { float d = t2[c] - mu; var += d * d; }
  var *= (1.f / 64.f);
  float rs = rsqrtf(var + LN_EPS);
#pragma unroll
  for (int c = 0; c < 64; ++c)
    h[(size_t)n * 64 + c] = (t2[c] - mu) * rs * sg[c] + sbt[c];
}

// ---------------- edge encoder fused with per-layer a_edge ---------------------
__global__ void edge_encoder(const float* __restrict__ ea,
                             const float* __restrict__ w1, const float* __restrict__ b1,
                             const float* __restrict__ w2, const float* __restrict__ b2,
                             const float* __restrict__ g,  const float* __restrict__ beta,
                             const float* __restrict__ Aew,
                             float* __restrict__ a_edge_all) {
  __shared__ float sw1[4 * 16], sb1[16], sw2[16 * 32], sb2[32], sg[32], sbt[32];
  __shared__ __align__(16) float saw[NLAYERS * EHID * HEADS];   // 384
  int tx = threadIdx.x;
  if (tx < 64)  sw1[tx] = w1[tx];
  if (tx < 16)  sb1[tx] = b1[tx];
  for (int t = tx; t < 512; t += 256) sw2[t] = w2[t];
  if (tx < 32) { sb2[tx] = b2[tx]; sg[tx] = g[tx]; sbt[tx] = beta[tx]; }
  for (int t = tx; t < NLAYERS * EHID * HEADS; t += 256) saw[t] = Aew[t];
  __syncthreads();
  int e = blockIdx.x * 256 + tx;
  if (e >= NE) return;
  float4 a4 = *(const float4*)(ea + (size_t)e * 4);
  float xi[4] = {a4.x, a4.y, a4.z, a4.w};
  float t1[16];
#pragma unroll
  for (int k = 0; k < 16; ++k) {
    float a = sb1[k];
#pragma unroll
    for (int j = 0; j < 4; ++j) a += xi[j] * sw1[j * 16 + k];
    t1[k] = a > 0.f ? a : 0.f;
  }
  float t2[32];
#pragma unroll
  for (int c = 0; c < 32; ++c) {
    float a = sb2[c];
#pragma unroll
    for (int k = 0; k < 16; ++k) a += t1[k] * sw2[k * 32 + c];
    t2[c] = a;
  }
  float mu = 0.f;
#pragma unroll
  for (int c = 0; c < 32; ++c) mu += t2[c];
  mu *= (1.f / 32.f);
  float var = 0.f;
#pragma unroll
  for (int c = 0; c < 32; ++c) { float d = t2[c] - mu; var += d * d; }
  var *= (1.f / 32.f);
  float rs = rsqrtf(var + LN_EPS);
  float ef[32];
#pragma unroll
  for (int c = 0; c < 32; ++c) ef[c] = (t2[c] - mu) * rs * sg[c] + sbt[c];
#pragma unroll
  for (int i = 0; i < NLAYERS; ++i) {
    float ax = 0.f, ay = 0.f, az = 0.f, aw = 0.f;
#pragma unroll
    for (int c = 0; c < EHID; ++c) {
      const float* p = &saw[(i * EHID + c) * 4];
      float v = ef[c];
      ax += v * p[0]; ay += v * p[1]; az += v * p[2]; aw += v * p[3];
    }
    *(float4*)(a_edge_all + ((size_t)i * NE + e) * 4) = make_float4(ax, ay, az, aw);
  }
}

// ---------------- per-layer node prep: h -> f16, a_src, a_dst ------------------
__global__ void node_prep(const float* __restrict__ h,
                          const float* __restrict__ Asw, const float* __restrict__ Adw,
                          _Float16* __restrict__ hf,
                          float* __restrict__ a_src, float* __restrict__ a_dst) {
  __shared__ float4 s_asw[HID], s_adw[HID];
  for (int t = threadIdx.x; t < HID; t += 256) {
    s_asw[t] = ((const float4*)Asw)[t];
    s_adw[t] = ((const float4*)Adw)[t];
  }
  __syncthreads();
  int n = blockIdx.x * 256 + threadIdx.x;
  if (n >= NN) return;
  const float4* hr = (const float4*)(h + (size_t)n * HID);
  _Float16* ho = hf + (size_t)n * HID;
  float sx = 0.f, sy = 0.f, sz = 0.f, sw = 0.f;
  float dx = 0.f, dy = 0.f, dz = 0.f, dw = 0.f;
#pragma unroll
  for (int q = 0; q < 16; ++q) {
    float4 v = hr[q];
    int k0 = q * 4;
    float4 a0 = s_asw[k0 + 0], a1 = s_asw[k0 + 1], a2 = s_asw[k0 + 2], a3 = s_asw[k0 + 3];
    float4 b0 = s_adw[k0 + 0], b1 = s_adw[k0 + 1], b2 = s_adw[k0 + 2], b3 = s_adw[k0 + 3];
    sx += v.x * a0.x + v.y * a1.x + v.z * a2.x + v.w * a3.x;
    sy += v.x * a0.y + v.y * a1.y + v.z * a2.y + v.w * a3.y;
    sz += v.x * a0.z + v.y * a1.z + v.z * a2.z + v.w * a3.z;
    sw += v.x * a0.w + v.y * a1.w + v.z * a2.w + v.w * a3.w;
    dx += v.x * b0.x + v.y * b1.x + v.z * b2.x + v.w * b3.x;
    dy += v.x * b0.y + v.y * b1.y + v.z * b2.y + v.w * b3.y;
    dz += v.x * b0.z + v.y * b1.z + v.z * b2.z + v.w * b3.z;
    dw += v.x * b0.w + v.y * b1.w + v.z * b2.w + v.w * b3.w;
    v4h pv = {(_Float16)v.x, (_Float16)v.y, (_Float16)v.z, (_Float16)v.w};
    *(v4h*)(ho + k0) = pv;
  }
  *(float4*)(a_src + (size_t)n * 4) = make_float4(sx, sy, sz, sw);
  *(float4*)(a_dst + (size_t)n * 4) = make_float4(dx, dy, dz, dw);
}

// ---------------- edge passes: leaky-relu + seg-max, exp + seg-sum, scatter ----
__global__ void edge_pass1(const int* __restrict__ ei,
                           const float* __restrict__ a_src, const float* __restrict__ a_dst,
                           const float* __restrict__ a_edge,
                           float* __restrict__ alph, unsigned* __restrict__ m_u) {
  int e = blockIdx.x * 256 + threadIdx.x;
  if (e >= NE) return;
  int s = ei[e], d = ei[NE + e];
  float4 vs = *(const float4*)(a_src + (size_t)s * 4);
  float4 vd = *(const float4*)(a_dst + (size_t)d * 4);
  float4 ve = *(const float4*)(a_edge + (size_t)e * 4);
  float al[4] = {vs.x + vd.x + ve.x, vs.y + vd.y + ve.y,
                 vs.z + vd.z + ve.z, vs.w + vd.w + ve.w};
#pragma unroll
  for (int hd = 0; hd < 4; ++hd) {
    float a = al[hd];
    a = a > 0.f ? a : 0.2f * a;
    al[hd] = a;
    atomicMax(&m_u[(size_t)d * 4 + hd], fmap_u(a));
  }
  *(float4*)(alph + (size_t)e * 4) = make_float4(al[0], al[1], al[2], al[3]);
}

__global__ void edge_pass2(const int* __restrict__ ei, const unsigned* __restrict__ m_u,
                           float* __restrict__ alph, float* __restrict__ ssum) {
  int e = blockIdx.x * 256 + threadIdx.x;
  if (e >= NE) return;
  int d = ei[NE + e];
  float4 av = *(const float4*)(alph + (size_t)e * 4);
  float al[4] = {av.x, av.y, av.z, av.w};
  float ex[4];
#pragma unroll
  for (int hd = 0; hd < 4; ++hd) {
    float m = funmap_u(m_u[(size_t)d * 4 + hd]);
    ex[hd] = expf(al[hd] - m);
    atomicAdd(&ssum[(size_t)d * 4 + hd], ex[hd]);
  }
  *(float4*)(alph + (size_t)e * 4) = make_float4(ex[0], ex[1], ex[2], ex[3]);
}

__global__ void edge_pass3(const int* __restrict__ ei, const float* __restrict__ ex,
                           const float* __restrict__ ssum, const float* __restrict__ hw,
                           float* __restrict__ agg) {
  int t = blockIdx.x * 256 + threadIdx.x;
  if (t >= NE * 4) return;
  int e = t >> 2, hd = t & 3;
  int s = ei[e], d = ei[NE + e];
  float w = ex[(size_t)e * 4 + hd] / (ssum[(size_t)d * 4 + hd] + 1e-16f);
  const float4* hp = (const float4*)(hw + (size_t)s * 256 + hd * 64);
  float* ap = agg + (size_t)d * 256 + hd * 64;
#pragma unroll
  for (int q = 0; q < 16; ++q) {
    float4 v = hp[q];
    atomicAdd(ap + q * 4 + 0, v.x * w);
    atomicAdd(ap + q * 4 + 1, v.y * w);
    atomicAdd(ap + q * 4 + 2, v.z * w);
    atomicAdd(ap + q * 4 + 3, v.w * w);
  }
}

// ---------------- node finish: mean heads + bias, relu, +res, LN ---------------
__global__ void node_finish(const float* __restrict__ agg, const float* __restrict__ bias,
                            const float* __restrict__ hres,
                            const float* __restrict__ g, const float* __restrict__ beta,
                            float* __restrict__ hout) {
  __shared__ float sb[64], sg[64], sbt[64];
  if (threadIdx.x < 64) { sb[threadIdx.x] = bias[threadIdx.x];
                          sg[threadIdx.x] = g[threadIdx.x];
                          sbt[threadIdx.x] = beta[threadIdx.x]; }
  __syncthreads();
  int n = blockIdx.x * 256 + threadIdx.x;
  if (n >= NN) return;
  float acc[64];
#pragma unroll
  for (int c = 0; c < 64; ++c) acc[c] = 0.f;
  const float4* ar = (const float4*)(agg + (size_t)n * 256);
#pragma unroll
  for (int hd = 0; hd < 4; ++hd)
#pragma unroll
    for (int q = 0; q < 16; ++q) {
      float4 v = ar[hd * 16 + q];
      acc[q * 4 + 0] += v.x; acc[q * 4 + 1] += v.y;
      acc[q * 4 + 2] += v.z; acc[q * 4 + 3] += v.w;
    }
  const float4* hr = (const float4*)(hres + (size_t)n * 64);
  float val[64];
#pragma unroll
  for (int q = 0; q < 16; ++q) {
    float4 r = hr[q];
    float rr[4] = {r.x, r.y, r.z, r.w};
#pragma unroll
    for (int j = 0; j < 4; ++j) {
      float tv = acc[q * 4 + j] * 0.25f + sb[q * 4 + j];
      tv = tv > 0.f ? tv : 0.f;
      val[q * 4 + j] = tv + rr[j];
    }
  }
  float mu = 0.f;
#pragma unroll
  for (int c = 0; c < 64; ++c) mu += val[c];
  mu *= (1.f / 64.f);
  float var = 0.f;
#pragma unroll
  for (int c = 0; c < 64; ++c) { float d = val[c] - mu; var += d * d; }
  var *= (1.f / 64.f);
  float rs = rsqrtf(var + LN_EPS);
#pragma unroll
  for (int c = 0; c < 64; ++c)
    hout[(size_t)n * 64 + c] = (val[c] - mu) * rs * sg[c] + sbt[c];
}

// ---------------- small helpers for the output MLP -----------------------------
__global__ void conv_f16(const float* __restrict__ in, _Float16* __restrict__ out, int count) {
  int t = blockIdx.x * 256 + threadIdx.x;
  if (t < count) out[t] = (_Float16)in[t];
}

__global__ void bias_relu_f16_128(const float* __restrict__ in, const float* __restrict__ b,
                                  _Float16* __restrict__ out, int total) {
  int t = blockIdx.x * 256 + threadIdx.x;
  if (t >= total) return;
  float v = in[t] + b[t & 127];
  out[t] = (_Float16)(v > 0.f ? v : 0.f);
}

// wave-per-node LayerNorm(256) in place on d_out (adds out_b2 first)
__global__ void out_ln(float* __restrict__ y, const float* __restrict__ b2,
                       const float* __restrict__ g, const float* __restrict__ beta) {
  int lane = threadIdx.x & 31;
  int wave = threadIdx.x >> 5;
  int n = blockIdx.x * 8 + wave;
  if (n >= NN) return;
  float* row = y + (size_t)n * 256;
  int c0 = lane * 8;
  float v[8];
#pragma unroll
  for (int j = 0; j < 8; ++j) v[j] = row[c0 + j] + b2[c0 + j];
  float s = 0.f;
#pragma unroll
  for (int j = 0; j < 8; ++j) s += v[j];
  for (int o = 16; o > 0; o >>= 1) s += __shfl_xor(s, o, 32);
  float mu = s * (1.f / 256.f);
  float q = 0.f;
#pragma unroll
  for (int j = 0; j < 8; ++j) { float d = v[j] - mu; q += d * d; }
  for (int o = 16; o > 0; o >>= 1) q += __shfl_xor(q, o, 32);
  float rs = rsqrtf(q * (1.f / 256.f) + LN_EPS);
#pragma unroll
  for (int j = 0; j < 8; ++j)
    row[c0 + j] = (v[j] - mu) * rs * g[c0 + j] + beta[c0 + j];
}

// ================================ host side ====================================
extern "C" void kernel_launch(void* const* d_in, const int* in_sizes, int n_in,
                              void* d_out, int out_size, void* d_ws, size_t ws_size,
                              hipStream_t stream) {
  (void)in_sizes; (void)n_in; (void)out_size; (void)ws_size;
  const float* x         = (const float*)d_in[0];
  const float* edge_attr = (const float*)d_in[1];
  const int*   ei        = (const int*)  d_in[2];
  const float* ne_w1 = (const float*)d_in[3];
  const float* ne_b1 = (const float*)d_in[4];
  const float* ne_w2 = (const float*)d_in[5];
  const float* ne_b2 = (const float*)d_in[6];
  const float* ne_g  = (const float*)d_in[7];
  const float* ne_bt = (const float*)d_in[8];
  const float* ee_w1 = (const float*)d_in[9];
  const float* ee_b1 = (const float*)d_in[10];
  const float* ee_w2 = (const float*)d_in[11];
  const float* ee_b2 = (const float*)d_in[12];
  const float* ee_g  = (const float*)d_in[13];
  const float* ee_bt = (const float*)d_in[14];
  const float* gat_lin      = (const float*)d_in[15];
  const float* gat_lin_edge = (const float*)d_in[16];
  const float* gat_att_src  = (const float*)d_in[17];
  const float* gat_att_dst  = (const float*)d_in[18];
  const float* gat_att_edge = (const float*)d_in[19];
  const float* gat_bias = (const float*)d_in[20];
  const float* gat_ln_g = (const float*)d_in[21];
  const float* gat_ln_b = (const float*)d_in[22];
  const float* out_w1 = (const float*)d_in[23];
  const float* out_b1 = (const float*)d_in[24];
  const float* out_w2 = (const float*)d_in[25];
  const float* out_b2 = (const float*)d_in[26];
  const float* out_g  = (const float*)d_in[27];
  const float* out_bt = (const float*)d_in[28];

  char* base = (char*)d_ws;
  size_t off = 0;
  auto carve = [&](size_t bytes) -> void* {
    void* p = base + off;
    off = (off + bytes + 255) & ~(size_t)255;
    return p;
  };
  float*    h0         = (float*)   carve((size_t)NN * HID * 4);
  float*    h1         = (float*)   carve((size_t)NN * HID * 4);
  _Float16* hf16       = (_Float16*)carve((size_t)NN * HID * 2);
  float*    hw         = (float*)   carve((size_t)NN * 256 * 4);
  float*    a_src      = (float*)   carve((size_t)NN * 4 * 4);
  float*    a_dst      = (float*)   carve((size_t)NN * 4 * 4);
  float*    a_edge_all = (float*)   carve((size_t)NLAYERS * NE * 4 * 4);
  float*    alph       = (float*)   carve((size_t)NE * 4 * 4);
  unsigned* m_u        = (unsigned*)carve((size_t)NN * 4 * 4);
  float*    ssum       = (float*)   carve((size_t)NN * 4 * 4);
  float*    agg        = (float*)   carve((size_t)NN * 256 * 4);
  float*    Asw        = (float*)   carve((size_t)NLAYERS * HID * 4 * 4);
  float*    Adw        = (float*)   carve((size_t)NLAYERS * HID * 4 * 4);
  float*    Aew        = (float*)   carve((size_t)NLAYERS * EHID * 4 * 4);
  _Float16* lin_pk     = (_Float16*)carve((size_t)NLAYERS * 16384 * 2);
  _Float16* w1_pk      = (_Float16*)carve((size_t)8192 * 2);
  _Float16* w2_pk      = (_Float16*)carve((size_t)32768 * 2);
  // output-stage buffers reuse edge scratch (edges are done by then)
  float*    t1  = a_edge_all;          // N*128 f32 = 51.2MB <= 76.8MB
  _Float16* t1f = (_Float16*)alph;     // N*128 f16 = 25.6MB <= 25.6MB

  const int nodeBlocks = (NN + 255) / 256;       // 391
  const int edgeBlocks = NE / 256;               // 6250
  const int Mtiles = NN / 16;                    // 6250 exactly

  // ---- one-time prep ----
  precompute_att<<<1, 256, 0, stream>>>(gat_lin, gat_lin_edge, gat_att_src,
                                        gat_att_dst, gat_att_edge, Asw, Adw, Aew);
  for (int i = 0; i < NLAYERS; ++i)
    pack_b_f16<<<64, 256, 0, stream>>>(gat_lin + (size_t)i * 64 * 256,
                                       lin_pk + (size_t)i * 16384, 64, 256);
  pack_b_f16<<<32, 256, 0, stream>>>(out_w1, w1_pk, 64, 128);
  pack_b_f16<<<128, 256, 0, stream>>>(out_w2, w2_pk, 128, 256);

  node_encoder<<<nodeBlocks, 256, 0, stream>>>(x, ne_w1, ne_b1, ne_w2, ne_b2,
                                               ne_g, ne_bt, h0);
  edge_encoder<<<edgeBlocks, 256, 0, stream>>>(edge_attr, ee_w1, ee_b1, ee_w2, ee_b2,
                                               ee_g, ee_bt, Aew, a_edge_all);

  // ---- GAT layers ----
  float* hcur = h0;
  float* hnxt = h1;
  for (int i = 0; i < NLAYERS; ++i) {
    node_prep<<<nodeBlocks, 256, 0, stream>>>(hcur, Asw + (size_t)i * 256,
                                              Adw + (size_t)i * 256, hf16, a_src, a_dst);
    gemm_wmma_f16<<<(Mtiles * 4 + 7) / 8, 256, 0, stream>>>(
        hf16, lin_pk + (size_t)i * 16384, hw, Mtiles, 2, 16);
    hipMemsetAsync(m_u,  0, (size_t)NN * 16,   stream);
    hipMemsetAsync(ssum, 0, (size_t)NN * 16,   stream);
    hipMemsetAsync(agg,  0, (size_t)NN * 1024, stream);
    edge_pass1<<<edgeBlocks, 256, 0, stream>>>(ei, a_src, a_dst,
                                               a_edge_all + (size_t)i * NE * 4, alph, m_u);
    edge_pass2<<<edgeBlocks, 256, 0, stream>>>(ei, m_u, alph, ssum);
    edge_pass3<<<edgeBlocks * 4, 256, 0, stream>>>(ei, alph, ssum, hw, agg);
    node_finish<<<nodeBlocks, 256, 0, stream>>>(agg, gat_bias + (size_t)i * 64, hcur,
                                                gat_ln_g + (size_t)i * 64,
                                                gat_ln_b + (size_t)i * 64, hnxt);
    float* tmp = hcur; hcur = hnxt; hnxt = tmp;
  }

  // ---- output MLP: 64 -> 128 relu -> 256 -> LN ----
  conv_f16<<<(NN * HID + 255) / 256, 256, 0, stream>>>(hcur, hf16, NN * HID);
  gemm_wmma_f16<<<(Mtiles * 2 + 7) / 8, 256, 0, stream>>>(hf16, w1_pk, t1, Mtiles, 2, 8);
  bias_relu_f16_128<<<(NN * 128 + 255) / 256, 256, 0, stream>>>(t1, out_b1, t1f, NN * 128);
  gemm_wmma_f16<<<(Mtiles * 4 + 7) / 8, 256, 0, stream>>>(t1f, w2_pk, (float*)d_out,
                                                          Mtiles, 4, 16);
  out_ln<<<(NN + 7) / 8, 256, 0, stream>>>((float*)d_out, out_b2, out_g, out_bt);
}
